// LigerFusedLinearGRPOLoss_89421219103752
// MI455X (gfx1250) — matrix operand order repaired and checked by reading, not compiled, not run on hardware
//
#include <hip/hip_runtime.h>
#include <hip/hip_bf16.h>
#include <math.h>

// Problem constants (from reference)
#define BATCH 4
#define TDIM  512
#define HDIM  2048
#define VDIM  32000
#define NTOK  (BATCH * TDIM)      // 2048
#define BETA  0.1f

// GEMM tiling
#define M_TILE   128
#define N_TILE   256
#define K_STEP   64
#define NCHUNK   (VDIM / N_TILE)  // 125
#define GRID_M   (NTOK / M_TILE)  // 16

typedef __bf16 bf16x16 __attribute__((ext_vector_type(16)));
typedef __bf16 bf16x8  __attribute__((ext_vector_type(8)));
typedef float  v8f     __attribute__((ext_vector_type(8)));

union Frag16 {
    bf16x16 v;
    bf16x8  h[2];
};

__device__ __forceinline__ unsigned int pack_bf16x2(float a, float b) {
    // round-to-nearest-even f32 -> bf16, two packed into a dword (low = first)
    unsigned int ua = __float_as_uint(a);
    unsigned int ub = __float_as_uint(b);
    unsigned int ra = (ua + 0x7FFFu + ((ua >> 16) & 1u)) >> 16;
    unsigned int rb = (ub + 0x7FFFu + ((ub >> 16) & 1u)) >> 16;
    return (ra & 0xFFFFu) | (rb << 16);
}

__device__ __forceinline__ uint4 pack8(const float4 a, const float4 b) {
    uint4 r;
    r.x = pack_bf16x2(a.x, a.y);
    r.y = pack_bf16x2(a.z, a.w);
    r.z = pack_bf16x2(b.x, b.y);
    r.w = pack_bf16x2(b.z, b.w);
    return r;
}

// LDS layout: bf16 rows of 64 K + 8 pad = 72 bf16 (144 B, 16B-aligned)
//   A: 128 rows * 144 B = 18432 B
//   B: 256 rows * 144 B = 36864 B   (offset 18432)
#define AB_STRIDE_BF  72
#define AB_STRIDE_U32 36
#define B_LDS_OFF     18432
#define SMEM_BYTES    55296

// -----------------------------------------------------------------------------
// Kernel 1: fused bf16-WMMA GEMM tile + per-row softmax partials
// grid: (GRID_M, NCHUNK, 2)   block: 256 threads (8 waves)
// Each wave: 16 rows x full 256 cols -> 16 accumulators.
// partials[((model*NTOK + row)*NCHUNK + chunk)*3] = {rowMax, sumExp, rowSum}
// -----------------------------------------------------------------------------
__global__ __launch_bounds__(256)
void grpo_gemm_partials(const float* __restrict__ inputs,
                        const float* __restrict__ weight,
                        const float* __restrict__ ref_inputs,
                        const float* __restrict__ ref_weight,
                        float* __restrict__ partials) {
    __shared__ __align__(16) unsigned char smem[SMEM_BYTES];
    unsigned int* As = (unsigned int*)smem;                  // [128][36] dwords
    unsigned int* Bs = (unsigned int*)(smem + B_LDS_OFF);    // [256][36] dwords

    const int model = blockIdx.z;
    const float* __restrict__ X = model ? ref_inputs : inputs;
    const float* __restrict__ W = model ? ref_weight : weight;

    const int m0 = blockIdx.x * M_TILE;
    const int n0 = blockIdx.y * N_TILE;

    const int tid  = threadIdx.x;
    const int wave = tid >> 5;        // wave w owns rows 16w..16w+15
    const int lane = tid & 31;
    const int hi   = lane >> 4;       // 0: lanes 0-15, 1: lanes 16-31
    const int l15  = lane & 15;

    v8f acc[16] = {};

    const __bf16* Ab = (const __bf16*)smem;
    const __bf16* Bb = (const __bf16*)(smem + B_LDS_OFF);
    const int arow = wave * 16 + l15;

    // staging coordinates (fp32 -> bf16 on the fly)
    const int a_row = tid >> 1;            // 128 rows, 2 threads/row
    const int a_seg = (tid & 1) * 32;      // 32 f32 each
    const float* a_src_base = X + (size_t)(m0 + a_row) * HDIM + a_seg;
    const float* b_src_base = W + (size_t)(n0 + tid) * HDIM;
    unsigned int* a_dst = &As[a_row * AB_STRIDE_U32 + (a_seg >> 1)];
    unsigned int* b_dst = &Bs[tid * AB_STRIDE_U32];

    for (int kk = 0; kk < HDIM; kk += K_STEP) {
        __syncthreads();  // previous compute finished reading LDS

        // ---- stage A tile: 128 rows x 64 K ----
        {
            const float4* s4 = (const float4*)(a_src_base + kk);
            uint4* d4 = (uint4*)a_dst;
#pragma unroll
            for (int i = 0; i < 4; ++i)
                d4[i] = pack8(s4[2 * i], s4[2 * i + 1]);
        }
        // ---- stage B tile: 256 rows x 64 K ----
        {
            const float4* s4 = (const float4*)(b_src_base + kk);
            uint4* d4 = (uint4*)b_dst;
#pragma unroll
            for (int i = 0; i < 8; ++i)
                d4[i] = pack8(s4[2 * i], s4[2 * i + 1]);
        }
        // ---- prefetch next stage (global_prefetch_b8) ----
        if (kk + K_STEP < HDIM) {
            __builtin_prefetch(a_src_base + kk + K_STEP, 0, 0);
            __builtin_prefetch(b_src_base + kk + K_STEP, 0, 0);
        }
        __syncthreads();  // staged data visible

        // ---- two K=32 substeps ----
#pragma unroll
        for (int s = 0; s < 2; ++s) {
            // A fragment (16x32 bf16, CDNA5 layout):
            // lanes 0-15: K = {0..7,16..23}; lanes 16-31: K = {8..15,24..31}
            Frag16 fa;
            const __bf16* ap = Ab + arow * AB_STRIDE_BF + s * 32;
            fa.h[0] = *(const bf16x8*)(ap + hi * 8);
            fa.h[1] = *(const bf16x8*)(ap + 16 + hi * 8);

#pragma unroll
            for (int g = 0; g < 4; ++g) {
                Frag16 fb[4];
#pragma unroll
                for (int j = 0; j < 4; ++j) {
                    // B fragment: col = subtile*16 + l15;
                    // lanes 0-15: K={0..7,16..23}, lanes 16-31: K={8..15,24..31}
                    const __bf16* bp = Bb +
                        (size_t)((g * 4 + j) * 16 + l15) * AB_STRIDE_BF +
                        s * 32 + hi * 8;
                    fb[j].h[0] = *(const bf16x8*)bp;
                    fb[j].h[1] = *(const bf16x8*)(bp + 16);
                }
#pragma unroll
                for (int j = 0; j < 4; ++j)
                    acc[g * 4 + j] = __builtin_amdgcn_wmma_f32_16x16x32_bf16(
                        false, fa.v, false, fb[j].v, (short)0,
                        acc[g * 4 + j], false, false);
            }
        }
    }

    // ---- register epilogue: per-row max / sum-exp / sum over 256 cols ----
    // D layout: lanes 0-15: N=l15, VGPR r: M=r; lanes 16-31: M=8+r.
    // Wave row (16*wave + 8*hi + r) is spread over 16 lanes x 16 subtiles.
#pragma unroll
    for (int r = 0; r < 8; ++r) {
        float m = acc[0][r];
#pragma unroll
        for (int j = 1; j < 16; ++j) m = fmaxf(m, acc[j][r]);
#pragma unroll
        for (int msk = 8; msk; msk >>= 1)
            m = fmaxf(m, __shfl_xor(m, msk, 16));

        float s = 0.0f, rs = 0.0f;
#pragma unroll
        for (int j = 0; j < 16; ++j) {
            const float x = acc[j][r];
            s  += __expf(x - m);
            rs += x;
        }
#pragma unroll
        for (int msk = 8; msk; msk >>= 1) {
            s  += __shfl_xor(s, msk, 16);
            rs += __shfl_xor(rs, msk, 16);
        }

        if (l15 == 0) {
            const int row = m0 + wave * 16 + hi * 8 + r;
            float* P = partials +
                ((size_t)(model * NTOK + row) * NCHUNK + blockIdx.y) * 3;
            P[0] = m;
            P[1] = s;
            P[2] = rs;
        }
    }
}

// -----------------------------------------------------------------------------
// Kernel 2: target logit per token (fp32 dot, full precision)
// grid: (NTOK, 2)  block: 256
// -----------------------------------------------------------------------------
__global__ __launch_bounds__(256)
void grpo_target_logit(const float* __restrict__ inputs,
                       const float* __restrict__ weight,
                       const float* __restrict__ ref_inputs,
                       const float* __restrict__ ref_weight,
                       const int* __restrict__ ids,
                       float* __restrict__ tgt) {
    __shared__ float red[256];
    const int tok   = blockIdx.x;
    const int model = blockIdx.y;
    const float* __restrict__ X = model ? ref_inputs : inputs;
    const float* __restrict__ W = model ? ref_weight : weight;
    const size_t id = (size_t)ids[tok];

    float p = 0.0f;
    const float* xr = X + (size_t)tok * HDIM;
    const float* wr = W + id * HDIM;
    for (int h = threadIdx.x; h < HDIM; h += 256)
        p += xr[h] * wr[h];

    red[threadIdx.x] = p;
    __syncthreads();
    for (int s = 128; s > 0; s >>= 1) {
        if (threadIdx.x < s) red[threadIdx.x] += red[threadIdx.x + s];
        __syncthreads();
    }
    if (threadIdx.x == 0) tgt[model * NTOK + tok] = red[0];
}

// -----------------------------------------------------------------------------
// Kernel 3: combine partials -> LSE -> per-token logp/KL -> loss + metrics
// grid: 1 block, 1024 threads. d_out = {loss, mean_lp, std_lp, full_mean, kl}
// -----------------------------------------------------------------------------
__device__ __forceinline__ float block_sum_1024(float v, float* buf) {
    const int t = threadIdx.x;
    buf[t] = v;
    __syncthreads();
    for (int s = 512; s > 0; s >>= 1) {
        if (t < s) buf[t] += buf[t + s];
        __syncthreads();
    }
    const float r = buf[0];
    __syncthreads();
    return r;
}

__global__ __launch_bounds__(1024)
void grpo_finalize(const float* __restrict__ partials,
                   const float* __restrict__ tgt,
                   const int* __restrict__ amask,
                   const float* __restrict__ adv,
                   float* __restrict__ out) {
    __shared__ float red[1024];

    float loss_s = 0.f, lp_s = 0.f, lp2_s = 0.f, full_s = 0.f,
          kl_s = 0.f, mask_s = 0.f;

    for (int t = threadIdx.x; t < NTOK; t += 1024) {
        float lp[2];
        float lse0 = 0.f, rsum0 = 0.f;
#pragma unroll
        for (int model = 0; model < 2; ++model) {
            const float* P = partials + (size_t)(model * NTOK + t) * NCHUNK * 3;
            float m = -3.4e38f;
            for (int c = 0; c < NCHUNK; ++c) m = fmaxf(m, P[3 * c]);
            float S = 0.f, rs = 0.f;
            for (int c = 0; c < NCHUNK; ++c) {
                S  += P[3 * c + 1] * __expf(P[3 * c] - m);
                rs += P[3 * c + 2];
            }
            const float lse = m + __logf(S);
            lp[model] = tgt[model * NTOK + t] - lse;
            if (model == 0) { lse0 = lse; rsum0 = rs; }
        }
        const float d  = lp[1] - lp[0];              // ref - policy
        const float kl = __expf(d) - d - 1.0f;
        const float mk = (float)amask[t];
        const float av = adv[t >> 9];                // T = 512
        const float per_token_loss = -(av - BETA * kl);  // ratio == 1

        loss_s += per_token_loss * mk;
        kl_s   += kl * mk;
        mask_s += mk;
        lp_s   += lp[0];
        lp2_s  += lp[0] * lp[0];
        full_s += rsum0 / (float)VDIM - lse0;
    }

    const float loss_t = block_sum_1024(loss_s, red);
    const float lp_t   = block_sum_1024(lp_s, red);
    const float lp2_t  = block_sum_1024(lp2_s, red);
    const float full_t = block_sum_1024(full_s, red);
    const float kl_t   = block_sum_1024(kl_s, red);
    const float mask_t = block_sum_1024(mask_s, red);

    if (threadIdx.x == 0) {
        const float N = (float)NTOK;
        out[0] = loss_t / (mask_t + 1e-8f);
        out[1] = lp_t / N;
        const float var = fmaxf(0.f, (lp2_t - lp_t * lp_t / N)) / (N - 1.0f);
        out[2] = sqrtf(var);
        out[3] = full_t / N;
        out[4] = kl_t / mask_t;
    }
}

// -----------------------------------------------------------------------------
extern "C" void kernel_launch(void* const* d_in, const int* in_sizes, int n_in,
                              void* d_out, int out_size, void* d_ws, size_t ws_size,
                              hipStream_t stream) {
    const float* weight     = (const float*)d_in[0];
    const float* inputs     = (const float*)d_in[1];
    const int*   input_ids  = (const int*)d_in[2];
    const int*   attn_mask  = (const int*)d_in[3];
    const float* advantages = (const float*)d_in[4];
    const float* ref_inputs = (const float*)d_in[5];
    const float* ref_weight = (const float*)d_in[6];

    float* tgt      = (float*)d_ws;                 // 2*NTOK floats
    float* partials = tgt + 2 * NTOK;               // 2*NTOK*NCHUNK*3 floats

    dim3 gGemm(GRID_M, NCHUNK, 2);
    grpo_gemm_partials<<<gGemm, 256, 0, stream>>>(
        inputs, weight, ref_inputs, ref_weight, partials);

    dim3 gTgt(NTOK, 2);
    grpo_target_logit<<<gTgt, 256, 0, stream>>>(
        inputs, weight, ref_inputs, ref_weight, input_ids, tgt);

    grpo_finalize<<<1, 1024, 0, stream>>>(
        partials, tgt, attn_mask, advantages, (float*)d_out);
}
